// MetaGraphNet_38757784879149
// MI455X (gfx1250) — matrix operand order repaired and checked by reading, not compiled, not run on hardware
//
#include <hip/hip_runtime.h>

// ---------------------------------------------------------------------------
// MetaGraphNet block for MI455X (gfx1250, wave32, WMMA).
// GEMMs: v_wmma_f32_16x16x32_bf16, barrier-free main loop (A/B fragments
// loaded straight from global/WGP$), fused GroupNorm/ReLU/residual epilogues.
// Edge gather stage uses GLOBAL_LOAD_ASYNC_TO_LDS (ASYNCcnt) for mem->LDS.
// ---------------------------------------------------------------------------

#define NN   2048
#define NE   16384
#define CHN  256
#define NHEADS 4
#define EPSN 1e-5f

typedef __bf16 bf16_t;
typedef __attribute__((ext_vector_type(16))) __bf16 v16bf;
typedef __attribute__((ext_vector_type(8)))  float  v8f;

enum { EPI_PLAIN = 0, EPI_GN_RELU = 1, EPI_RES = 2 };

// ------------------------- helpers -----------------------------------------

__device__ inline void atomicMaxFloat(float* addr, float val) {
  int* ia = (int*)addr;
  int old = __float_as_int(*addr);
  while (__int_as_float(old) < val) {
    int assumed = old;
    old = atomicCAS(ia, assumed, __float_as_int(val));
    if (old == assumed) break;
  }
}

// Convert weight [K, 256] f32 row-major -> [256, K] bf16 row-major (W^T),
// so each WMMA B-fragment lane reads contiguous K-pairs.
__global__ __launch_bounds__(256)
void wconv_kernel(const float* __restrict__ W, bf16_t* __restrict__ Wt, int K) {
  int i = blockIdx.x * 256 + threadIdx.x;
  if (i >= K * CHN) return;
  int k = i / CHN, n = i % CHN;
  Wt[(size_t)n * K + k] = (bf16_t)W[i];
}

__global__ __launch_bounds__(256)
void f2b_kernel(const float* __restrict__ s, bf16_t* __restrict__ d, int n) {
  int i = blockIdx.x * 256 + threadIdx.x;
  if (i < n) d[i] = (bf16_t)s[i];
}

// ------------------------- edge concat + GN(3ch, groups of 24) + ReLU -------
// Stages [x[row] | x[col] | edge_attr] into LDS via async mem->LDS copies
// (GLOBAL_LOAD_ASYNC_TO_LDS_B32, tracked by ASYNCcnt), then GroupNorm+ReLU.

__global__ __launch_bounds__(256)
void edge_gather_gn_kernel(const float* __restrict__ x, const int* __restrict__ ei,
                           const float* __restrict__ ea,
                           const float* __restrict__ gamma, const float* __restrict__ beta,
                           bf16_t* __restrict__ outp /* [E, 768] bf16 */) {
  int e = blockIdx.x, t = threadIdx.x;
  __shared__ float h[3 * CHN];
  __shared__ float mu[32], rs[32];
  int r = ei[e], c = ei[NE + e];
  {
    // per-lane LDS byte address (low 32 bits of flat shared pointer) and
    // per-lane global sources; 3 async B32 copies per lane -> 3KB tile.
    unsigned l0 = (unsigned)(size_t)(&h[0]) + (unsigned)(t * 4);
    const float* g0 = x + (size_t)r * CHN + t;
    const float* g1 = x + (size_t)c * CHN + t;
    const float* g2 = ea + (size_t)e * CHN + t;
    asm volatile("global_load_async_to_lds_b32 %0, %1, off"
                 :: "v"(l0), "v"(g0) : "memory");
    asm volatile("global_load_async_to_lds_b32 %0, %1, off"
                 :: "v"(l0 + 1024u), "v"(g1) : "memory");
    asm volatile("global_load_async_to_lds_b32 %0, %1, off"
                 :: "v"(l0 + 2048u), "v"(g2) : "memory");
    asm volatile("s_wait_asynccnt 0x0" ::: "memory");
  }
  __syncthreads();
  if (t < 32) {  // 32 groups x 24 channels
    float s = 0.f, ss = 0.f;
    for (int i = 0; i < 24; ++i) { float v = h[t * 24 + i]; s += v; ss += v * v; }
    float m = s * (1.f / 24.f);
    mu[t] = m; rs[t] = rsqrtf(ss * (1.f / 24.f) - m * m + EPSN);
  }
  __syncthreads();
  for (int cc = t; cc < 3 * CHN; cc += 256) {
    int g = cc / 24;
    float v = (h[cc] - mu[g]) * rs[g] * gamma[cc] + beta[cc];
    outp[(size_t)e * (3 * CHN) + cc] = (bf16_t)fmaxf(v, 0.f);
  }
}

// ------------------------- generic bf16 WMMA GEMM, N = 256 ------------------
// Block: 256 threads (8 wave32s) -> 32 rows x 256 cols of C.
// Each wave: 2 M-tiles x 2 N-tiles = 4 x v_wmma_f32_16x16x32_bf16 per 32-K
// step; A and B fragments are contiguous 16B chunks loaded straight from
// global (A rows shared by all waves hit WGP$) -> no LDS / barriers in loop.
// A: [M, K] bf16 row-major.  Bt: [256, K] bf16 row-major (transposed weight).

#define WMMA_BF16(A_, B_, C_) \
  __builtin_amdgcn_wmma_f32_16x16x32_bf16(false, (A_), false, (B_), (short)0, (C_), false, false)

__global__ __launch_bounds__(256)
void gemm256_kernel(const bf16_t* __restrict__ A, int K,
                    const bf16_t* __restrict__ Bt,
                    const float* __restrict__ bias,
                    int epi,
                    const float* __restrict__ gamma,
                    const float* __restrict__ beta,
                    const float* __restrict__ resid,
                    float* __restrict__ outF,
                    bf16_t* __restrict__ outB) {
  __shared__ float Csm[32][CHN + 4];
  const int t = threadIdx.x;
  const int wave = t >> 5, lane = t & 31;
  const int laneQ = lane & 15, hiHalf = lane >> 4;
  const int m0 = blockIdx.x * 32;
  const int n0 = wave * 32;
  v8f c00 = {}, c01 = {}, c10 = {}, c11 = {};

  // A fragment rows (ISA A layout: lanes 0-15 -> K{0..7,16..23}; 16-31 -> K{8..15,24..31})
  const bf16_t* arow0 = A + (size_t)(m0 + laneQ) * K;
  const bf16_t* arow1 = arow0 + (size_t)16 * K;
  // B fragment rows (lane = column of B = row of Wt; same K layout as A)
  const bf16_t* brow0 = Bt + (size_t)(n0 + laneQ) * K;
  const bf16_t* brow1 = brow0 + (size_t)16 * K;
  const int kA = hiHalf * 8;
  const int kB = hiHalf * 8;

  for (int k0 = 0; k0 < K; k0 += 32) {
    union { v16bf v; uint4 q[2]; } a0, a1, b0, b1;
    a0.q[0] = *(const uint4*)(arow0 + k0 + kA);
    a0.q[1] = *(const uint4*)(arow0 + k0 + kA + 16);
    a1.q[0] = *(const uint4*)(arow1 + k0 + kA);
    a1.q[1] = *(const uint4*)(arow1 + k0 + kA + 16);
    b0.q[0] = *(const uint4*)(brow0 + k0 + kB);
    b0.q[1] = *(const uint4*)(brow0 + k0 + kB + 16);
    b1.q[0] = *(const uint4*)(brow1 + k0 + kB);
    b1.q[1] = *(const uint4*)(brow1 + k0 + kB + 16);
    c00 = WMMA_BF16(a0.v, b0.v, c00);
    c01 = WMMA_BF16(a0.v, b1.v, c01);
    c10 = WMMA_BF16(a1.v, b0.v, c10);
    c11 = WMMA_BF16(a1.v, b1.v, c11);
  }

  {  // spill accumulators (+bias) to LDS; C layout: lane<16 -> N=lane, M=vgpr; lane>=16 -> M=8+vgpr
    int col0 = n0 + laneQ, col1 = col0 + 16;
    int rb = hiHalf * 8;
    float bb0 = bias[col0], bb1 = bias[col1];
#pragma unroll
    for (int i = 0; i < 8; ++i) {
      Csm[rb + i][col0]      = c00[i] + bb0;
      Csm[rb + i][col1]      = c01[i] + bb1;
      Csm[16 + rb + i][col0] = c10[i] + bb0;
      Csm[16 + rb + i][col1] = c11[i] + bb1;
    }
  }
  __syncthreads();

  if (epi == EPI_GN_RELU) {  // per-row GroupNorm, 32 groups of 8 channels, + ReLU -> bf16
    int g0 = t * 4;          // 32 rows x 32 groups = 1024 groups / 256 threads
    for (int g = g0; g < g0 + 4; ++g) {
      int r = g >> 5, cb = (g & 31) * 8;
      float s = 0.f, ss = 0.f;
#pragma unroll
      for (int i = 0; i < 8; ++i) { float v = Csm[r][cb + i]; s += v; ss += v * v; }
      float m = s * 0.125f;
      float rstd = rsqrtf(ss * 0.125f - m * m + EPSN);
#pragma unroll
      for (int i = 0; i < 8; ++i) {
        int c = cb + i;
        float v = (Csm[r][c] - m) * rstd * gamma[c] + beta[c];
        outB[(size_t)(m0 + r) * CHN + c] = (bf16_t)fmaxf(v, 0.f);
      }
    }
  } else if (epi == EPI_RES) {  // + residual -> f32 out (and optional bf16 copy)
    for (int r = 0; r < 32; ++r) {
      float v = Csm[r][t] + resid[(size_t)(m0 + r) * CHN + t];
      outF[(size_t)(m0 + r) * CHN + t] = v;
      if (outB) outB[(size_t)(m0 + r) * CHN + t] = (bf16_t)v;
    }
  } else {  // plain f32 store
    for (int r = 0; r < 32; ++r)
      outF[(size_t)(m0 + r) * CHN + t] = Csm[r][t];
  }
}

// ------------------------- sparse attention ---------------------------------

__global__ __launch_bounds__(256)
void attn_init_kernel(float* mmax, float* den, float* gacc) {
  int i = blockIdx.x * 256 + threadIdx.x;
  if (i < NN * NHEADS) { mmax[i] = -1e30f; den[i] = 0.f; }
  if (i < NN * CHN) gacc[i] = 0.f;
}

__global__ __launch_bounds__(256)
void attn_score_kernel(const float* __restrict__ q, const float* __restrict__ k,
                       const int* __restrict__ ei, float* __restrict__ sarr,
                       float* __restrict__ mmax) {
  int idx = blockIdx.x * 256 + threadIdx.x;
  if (idx >= NE * NHEADS) return;
  int e = idx >> 2, h = idx & 3;
  int n = ei[NE + e];  // destination node (col)
  const float* qp = q + (size_t)n * CHN + h * 64;
  const float* kp = k + (size_t)e * CHN + h * 64;
  float s = 0.f;
#pragma unroll 8
  for (int d = 0; d < 64; ++d) s += qp[d] * kp[d];
  s *= 0.125f;  // 1/sqrt(d_k), d_k = 64
  sarr[idx] = s;
  atomicMaxFloat(&mmax[n * NHEADS + h], s);
}

__global__ __launch_bounds__(256)
void attn_expsum_kernel(float* __restrict__ sarr, const float* __restrict__ mmax,
                        const int* __restrict__ ei, float* __restrict__ den) {
  int idx = blockIdx.x * 256 + threadIdx.x;
  if (idx >= NE * NHEADS) return;
  int e = idx >> 2, h = idx & 3;
  int n = ei[NE + e];
  float w = expf(sarr[idx] - mmax[n * NHEADS + h]);
  sarr[idx] = w;
  atomicAdd(&den[n * NHEADS + h], w);
}

__global__ __launch_bounds__(256)
void vmean_kernel(const float* __restrict__ v, float* __restrict__ vmean) {
  int c = blockIdx.x, t = threadIdx.x;
  __shared__ float red[256];
  float s = 0.f;
  for (int e = t; e < NE; e += 256) s += v[(size_t)e * CHN + c];
  red[t] = s; __syncthreads();
  for (int o = 128; o; o >>= 1) { if (t < o) red[t] += red[t + o]; __syncthreads(); }
  if (t == 0) vmean[c] = red[0] * (1.f / NE);
}

__global__ __launch_bounds__(256)
void attn_scatter_kernel(const float* __restrict__ sarr, const float* __restrict__ v,
                         const int* __restrict__ ei, float* __restrict__ gacc) {
  int idx = blockIdx.x * 256 + threadIdx.x;
  if (idx >= NE * CHN) return;
  int e = idx >> 8, c = idx & 255, h = c >> 6;
  int n = ei[NE + e];
  atomicAdd(&gacc[(size_t)n * CHN + c], sarr[e * NHEADS + h] * v[idx]);
}

// Zero-in-degree node in dense reference => uniform softmax over ALL edges => mean(v).
__global__ __launch_bounds__(256)
void attn_final_kernel(const float* __restrict__ gacc, const float* __restrict__ den,
                       const float* __restrict__ vmean, bf16_t* __restrict__ gb) {
  int idx = blockIdx.x * 256 + threadIdx.x;
  if (idx >= NN * CHN) return;
  int n = idx >> 8, c = idx & 255, h = c >> 6;
  float d = den[n * NHEADS + h];
  float g = (d > 0.f) ? gacc[idx] / d : vmean[c];
  gb[idx] = (bf16_t)g;
}

// ------------------------- node concat: [GN(x), g] -> bf16 ------------------

__global__ __launch_bounds__(256)
void node_cat_kernel(const float* __restrict__ x, const float* __restrict__ g,
                     const float* __restrict__ gamma, const float* __restrict__ beta,
                     bf16_t* __restrict__ outp /* [N, 512] */) {
  int n = blockIdx.x, t = threadIdx.x;
  __shared__ float h[CHN];
  __shared__ float mu[32], rs[32];
  h[t] = x[(size_t)n * CHN + t];
  __syncthreads();
  if (t < 32) {
    float s = 0.f, ss = 0.f;
    for (int i = 0; i < 8; ++i) { float v = h[t * 8 + i]; s += v; ss += v * v; }
    float m = s * 0.125f;
    mu[t] = m; rs[t] = rsqrtf(ss * 0.125f - m * m + EPSN);
  }
  __syncthreads();
  int grp = t >> 3;
  float xa = (h[t] - mu[grp]) * rs[grp] * gamma[t] + beta[t];
  outp[(size_t)n * (2 * CHN) + t] = (bf16_t)xa;
  outp[(size_t)n * (2 * CHN) + CHN + t] = (bf16_t)g[(size_t)n * CHN + t];
}

// ------------------------- launcher -----------------------------------------

extern "C" void kernel_launch(void* const* d_in, const int* in_sizes, int n_in,
                              void* d_out, int out_size, void* d_ws, size_t ws_size,
                              hipStream_t stream) {
  const float* x    = (const float*)d_in[0];
  const int*   ei   = (const int*)d_in[1];
  const float* ea   = (const float*)d_in[2];
  const float* gE0g = (const float*)d_in[3];
  const float* gE0b = (const float*)d_in[4];
  const float* We1  = (const float*)d_in[5];
  const float* be1  = (const float*)d_in[6];
  const float* gE1g = (const float*)d_in[7];
  const float* gE1b = (const float*)d_in[8];
  const float* We2  = (const float*)d_in[9];
  const float* be2  = (const float*)d_in[10];
  const float* Wq   = (const float*)d_in[11];
  const float* bq   = (const float*)d_in[12];
  const float* Wk   = (const float*)d_in[13];
  const float* bk   = (const float*)d_in[14];
  const float* Wv   = (const float*)d_in[15];
  const float* bv   = (const float*)d_in[16];
  const float* Wo   = (const float*)d_in[17];
  const float* bo   = (const float*)d_in[18];
  const float* gNg  = (const float*)d_in[19];
  const float* gNb  = (const float*)d_in[20];
  const float* Wn1  = (const float*)d_in[21];
  const float* bn1  = (const float*)d_in[22];
  const float* gN1g = (const float*)d_in[23];
  const float* gN1b = (const float*)d_in[24];
  const float* Wn2  = (const float*)d_in[25];
  const float* bn2  = (const float*)d_in[26];
  float* out = (float*)d_out;

  // workspace layout (~73 MB, 256B-aligned slots)
  char* wsb = (char*)d_ws; size_t off = 0;
  auto alloc = [&](size_t bytes) -> void* {
    void* p = wsb + off; off += (bytes + 255) & ~(size_t)255; return p;
  };
  bf16_t* We1t = (bf16_t*)alloc((size_t)CHN * 768 * 2);
  bf16_t* We2t = (bf16_t*)alloc((size_t)CHN * 256 * 2);
  bf16_t* Wqt  = (bf16_t*)alloc((size_t)CHN * 256 * 2);
  bf16_t* Wkt  = (bf16_t*)alloc((size_t)CHN * 256 * 2);
  bf16_t* Wvt  = (bf16_t*)alloc((size_t)CHN * 256 * 2);
  bf16_t* Wot  = (bf16_t*)alloc((size_t)CHN * 256 * 2);
  bf16_t* Wn1t = (bf16_t*)alloc((size_t)CHN * 512 * 2);
  bf16_t* Wn2t = (bf16_t*)alloc((size_t)CHN * 256 * 2);
  bf16_t* xb   = (bf16_t*)alloc((size_t)NN * CHN * 2);
  bf16_t* hA   = (bf16_t*)alloc((size_t)NE * 768 * 2);   // edge concat (25 MB)
  bf16_t* h1b  = (bf16_t*)alloc((size_t)NE * CHN * 2);
  bf16_t* enb  = (bf16_t*)alloc((size_t)NE * CHN * 2);   // e_new bf16
  float*  qf   = (float*)alloc((size_t)NN * CHN * 4);
  float*  kf   = (float*)alloc((size_t)NE * CHN * 4);
  float*  vf   = (float*)hA;                              // reuse: hA dead after edge GEMM1
  float*  sarr = (float*)alloc((size_t)NE * NHEADS * 4);
  float*  mmax = (float*)alloc((size_t)NN * NHEADS * 4);
  float*  den  = (float*)alloc((size_t)NN * NHEADS * 4);
  float*  gacc = (float*)alloc((size_t)NN * CHN * 4);
  float*  vme  = (float*)alloc((size_t)CHN * 4);
  bf16_t* gb   = (bf16_t*)alloc((size_t)NN * CHN * 2);
  float*  gout = (float*)alloc((size_t)NN * CHN * 4);
  bf16_t* hcat = (bf16_t*)alloc((size_t)NN * 512 * 2);
  bf16_t* hn1b = (bf16_t*)alloc((size_t)NN * CHN * 2);

  // weight prep (bf16, transposed)
  auto wcv = [&](const float* W, bf16_t* Wt, int K) {
    wconv_kernel<<<(K * CHN + 255) / 256, 256, 0, stream>>>(W, Wt, K);
  };
  wcv(We1, We1t, 768); wcv(We2, We2t, 256);
  wcv(Wq, Wqt, 256);   wcv(Wk, Wkt, 256);
  wcv(Wv, Wvt, 256);   wcv(Wo, Wot, 256);
  wcv(Wn1, Wn1t, 512); wcv(Wn2, Wn2t, 256);
  f2b_kernel<<<(NN * CHN + 255) / 256, 256, 0, stream>>>(x, xb, NN * CHN);

  // ---- edge model ----
  edge_gather_gn_kernel<<<NE, 256, 0, stream>>>(x, ei, ea, gE0g, gE0b, hA);
  gemm256_kernel<<<NE / 32, 256, 0, stream>>>(hA, 768, We1t, be1, EPI_GN_RELU,
                                              gE1g, gE1b, nullptr, nullptr, h1b);
  gemm256_kernel<<<NE / 32, 256, 0, stream>>>(h1b, 256, We2t, be2, EPI_RES,
                                              nullptr, nullptr, ea,
                                              out + (size_t)NN * CHN, enb);

  // ---- projections ----
  gemm256_kernel<<<NN / 32, 256, 0, stream>>>(xb, 256, Wqt, bq, EPI_PLAIN,
                                              nullptr, nullptr, nullptr, qf, nullptr);
  gemm256_kernel<<<NE / 32, 256, 0, stream>>>(enb, 256, Wkt, bk, EPI_PLAIN,
                                              nullptr, nullptr, nullptr, kf, nullptr);
  gemm256_kernel<<<NE / 32, 256, 0, stream>>>(enb, 256, Wvt, bv, EPI_PLAIN,
                                              nullptr, nullptr, nullptr, vf, nullptr);

  // ---- sparse multi-head attention over incoming edges ----
  attn_init_kernel<<<(NN * CHN) / 256, 256, 0, stream>>>(mmax, den, gacc);
  attn_score_kernel<<<(NE * NHEADS) / 256, 256, 0, stream>>>(qf, kf, ei, sarr, mmax);
  attn_expsum_kernel<<<(NE * NHEADS) / 256, 256, 0, stream>>>(sarr, mmax, ei, den);
  vmean_kernel<<<CHN, 256, 0, stream>>>(vf, vme);
  attn_scatter_kernel<<<(NE * CHN) / 256, 256, 0, stream>>>(sarr, vf, ei, gacc);
  attn_final_kernel<<<(NN * CHN) / 256, 256, 0, stream>>>(gacc, den, vme, gb);
  gemm256_kernel<<<NN / 32, 256, 0, stream>>>(gb, 256, Wot, bo, EPI_PLAIN,
                                              nullptr, nullptr, nullptr, gout, nullptr);

  // ---- node model ----
  node_cat_kernel<<<NN, 256, 0, stream>>>(x, gout, gNg, gNb, hcat);
  gemm256_kernel<<<NN / 32, 256, 0, stream>>>(hcat, 512, Wn1t, bn1, EPI_GN_RELU,
                                              gN1g, gN1b, nullptr, nullptr, hn1b);
  gemm256_kernel<<<NN / 32, 256, 0, stream>>>(hn1b, 256, Wn2t, bn2, EPI_RES,
                                              nullptr, nullptr, x, out, nullptr);
}
